// EncoderBlock_23502061043962
// MI455X (gfx1250) — matrix operand order repaired and checked by reading, not compiled, not run on hardware
//
#include <hip/hip_runtime.h>
#include <hip/hip_bf16.h>
#include <stdint.h>

typedef __attribute__((ext_vector_type(16))) _Float16 v16h;
typedef __attribute__((ext_vector_type(8)))  _Float16 v8h;
typedef __attribute__((ext_vector_type(8)))  float    v8f;
typedef __attribute__((ext_vector_type(4)))  int      v4i;

#define S_LEN  4096
#define DMODEL 512
#define NHEAD  8
#define DKH    64
#define FFDIM  2048

#define WMMA_F16(a, b, c) \
  __builtin_amdgcn_wmma_f32_16x16x32_f16(false, (a), false, (b), (short)0, (c), false, false)

#ifndef __has_builtin
#define __has_builtin(x) 0
#endif

#if __has_builtin(__builtin_amdgcn_global_load_async_to_lds_b128)
#define HAVE_ASYNC_LDS 1
#else
#define HAVE_ASYNC_LDS 0
#endif

#define AS1 __attribute__((address_space(1)))
#define AS3 __attribute__((address_space(3)))

// Copy 16B global -> LDS. Async (ASYNCcnt) when available, sync fallback otherwise.
__device__ __forceinline__ void stage16(const _Float16* __restrict__ g, _Float16* l) {
#if HAVE_ASYNC_LDS
  // generic->AS1: flat global address == global address;
  // generic->AS3: LDS aperture keeps the byte offset in the low 32 bits (ISA §10.2).
  __builtin_amdgcn_global_load_async_to_lds_b128(
      (AS1 v4i*)(uintptr_t)g, (AS3 v4i*)(uint32_t)(uintptr_t)l, 0, 0);
#else
  *(v8h*)l = *(const v8h*)g;
#endif
}

__device__ __forceinline__ void wait_async_stage() {
#if HAVE_ASYNC_LDS
#if __has_builtin(__builtin_amdgcn_s_wait_asynccnt)
  __builtin_amdgcn_s_wait_asynccnt(0);
#else
  asm volatile("s_wait_asynccnt 0" ::: "memory");
#endif
#endif
}

// ---------------------------------------------------------------------------
// Fragment loaders (layouts per CDNA5 ISA §7.12.2)
// A 16-bit 16x32 (MxK): lanes 0-15 row=lane, K 0..7 & 16..23; lanes 16-31 row=lane-16, K 8..15 & 24..31
// ---------------------------------------------------------------------------
__device__ __forceinline__ v16h load_a_frag(const _Float16* __restrict__ base,
                                            int stride, int row0, int lane) {
  int l = lane & 15, hi = (lane >> 4) & 1;
  const _Float16* p = base + (size_t)(row0 + l) * stride + hi * 8;
  v8h x0 = *(const v8h*)(p);
  v8h x1 = *(const v8h*)(p + 16);
  v16h r;
#pragma unroll
  for (int i = 0; i < 8; ++i) { r[i] = x0[i]; r[8 + i] = x1[i]; }
  return r;
}

// B 16-bit 32x16 (KxN), stored transposed Bt[n][k]: lanes 0-15 col=lane hold K 0..7 & 16..23;
// lanes 16-31 col=lane-16 hold K 8..15 & 24..31 (same order as A).
__device__ __forceinline__ v16h load_b_frag(const _Float16* __restrict__ base,
                                            int stride, int col0, int lane) {
  int l = lane & 15, hi = (lane >> 4) & 1;
  const _Float16* p = base + (size_t)(col0 + l) * stride + hi * 8;
  v8h x0 = *(const v8h*)(p);
  v8h x1 = *(const v8h*)(p + 16);
  v16h r;
#pragma unroll
  for (int i = 0; i < 8; ++i) { r[i] = x0[i]; r[8 + i] = x1[i]; }
  return r;
}

// ---------------------------------------------------------------------------
// Weight transpose + f32->f16 cast:  wt[n*K + k] = w[k*N + n]
// ---------------------------------------------------------------------------
__global__ void transpose_cast_kernel(const float* __restrict__ w,
                                      _Float16* __restrict__ wt, int K, int N) {
  int idx = blockIdx.x * 256 + threadIdx.x;
  if (idx < K * N) {
    int n = idx / K, k = idx % K;
    wt[idx] = (_Float16)w[k * N + n];
  }
}

// ---------------------------------------------------------------------------
// LayerNorm (torch.std semantics: Bessel, eps added to std), f32 in, f16 out.
// ---------------------------------------------------------------------------
__global__ __launch_bounds__(256)
void ln_kernel(const float* __restrict__ x, const float* __restrict__ alpha,
               const float* __restrict__ beta, _Float16* __restrict__ outh) {
  __shared__ float red[256];
  int r = blockIdx.x, tid = threadIdx.x;
  const float* xr = x + (size_t)r * DMODEL;
  float v0 = xr[tid], v1 = xr[tid + 256];
  red[tid] = v0 + v1;
  __syncthreads();
#pragma unroll
  for (int off = 128; off > 0; off >>= 1) {
    if (tid < off) red[tid] += red[tid + off];
    __syncthreads();
  }
  float mean = red[0] * (1.0f / DMODEL);
  __syncthreads();
  float d0 = v0 - mean, d1 = v1 - mean;
  red[tid] = d0 * d0 + d1 * d1;
  __syncthreads();
#pragma unroll
  for (int off = 128; off > 0; off >>= 1) {
    if (tid < off) red[tid] += red[tid + off];
    __syncthreads();
  }
  float stdv = sqrtf(red[0] * (1.0f / (DMODEL - 1)));
  float inv = alpha[0] / (stdv + 1e-6f);
  float b = beta[0];
  outh[(size_t)r * DMODEL + tid]       = (_Float16)(d0 * inv + b);
  outh[(size_t)r * DMODEL + tid + 256] = (_Float16)(d1 * inv + b);
}

// ---------------------------------------------------------------------------
// WMMA GEMM: C[M,N] = act(A[M,K] @ Bt[N,K]^T + bias) (+ residual)
// Block tile 128x128, 8 waves of 32x64, K-step 32.
// Double-buffered LDS, async global->LDS staging when available.
// ---------------------------------------------------------------------------
__global__ __launch_bounds__(256)
void gemm_kernel(const _Float16* __restrict__ A, const _Float16* __restrict__ Bt,
                 const float* __restrict__ bias, const float* __restrict__ resid,
                 float* __restrict__ outf, _Float16* __restrict__ outh,
                 int M, int N, int K, int relu) {
  __shared__ __align__(16) _Float16 Asm[2][128 * 32];
  __shared__ __align__(16) _Float16 Bsm[2][128 * 32];
  int tid = threadIdx.x, lane = tid & 31, wave = tid >> 5;
  int wm = (wave >> 1) * 32;   // 0,32,64,96
  int wn = (wave & 1) * 64;    // 0,64
  int bm = blockIdx.y * 128;
  int bn = blockIdx.x * 128;

  // staging coordinates: 2 chunks of 16B per thread per tile
  int srow0 = tid >> 2,           skc0 = (tid & 3) * 8;
  int srow1 = (tid + 256) >> 2,   skc1 = ((tid + 256) & 3) * 8;

  // prologue: stage K-tile 0 into buffer 0
  stage16(&A[(size_t)(bm + srow0) * K + skc0], &Asm[0][srow0 * 32 + skc0]);
  stage16(&A[(size_t)(bm + srow1) * K + skc1], &Asm[0][srow1 * 32 + skc1]);
  stage16(&Bt[(size_t)(bn + srow0) * K + skc0], &Bsm[0][srow0 * 32 + skc0]);
  stage16(&Bt[(size_t)(bn + srow1) * K + skc1], &Bsm[0][srow1 * 32 + skc1]);
  wait_async_stage();
  __syncthreads();

  v8f acc[2][4] = {};
  int buf = 0;
  for (int k0 = 0; k0 < K; k0 += 32, buf ^= 1) {
    // stage next K-tile into the other buffer while we compute on this one
    if (k0 + 32 < K) {
      int kn = k0 + 32;
      stage16(&A[(size_t)(bm + srow0) * K + kn + skc0], &Asm[buf ^ 1][srow0 * 32 + skc0]);
      stage16(&A[(size_t)(bm + srow1) * K + kn + skc1], &Asm[buf ^ 1][srow1 * 32 + skc1]);
      stage16(&Bt[(size_t)(bn + srow0) * K + kn + skc0], &Bsm[buf ^ 1][srow0 * 32 + skc0]);
      stage16(&Bt[(size_t)(bn + srow1) * K + kn + skc1], &Bsm[buf ^ 1][srow1 * 32 + skc1]);
    }

    v16h a0 = load_a_frag(Asm[buf], 32, wm, lane);
    v16h a1 = load_a_frag(Asm[buf], 32, wm + 16, lane);
#pragma unroll
    for (int j = 0; j < 4; ++j) {
      v16h bf = load_b_frag(Bsm[buf], 32, wn + j * 16, lane);
      acc[0][j] = WMMA_F16(a0, bf, acc[0][j]);
      acc[1][j] = WMMA_F16(a1, bf, acc[1][j]);
    }

    wait_async_stage();   // next buffer fully written (our own contributions)
    __syncthreads();      // everyone done reading current buffer + all writes visible
  }

  int l = lane & 15, hi = (lane >> 4) & 1;
#pragma unroll
  for (int i = 0; i < 2; ++i) {
#pragma unroll
    for (int j = 0; j < 4; ++j) {
      int col = bn + wn + j * 16 + l;
      float bv = bias[col];
#pragma unroll
      for (int e = 0; e < 8; ++e) {
        int row = bm + wm + i * 16 + e + 8 * hi;
        float val = acc[i][j][e] + bv;
        if (relu) val = fmaxf(val, 0.0f);
        size_t o = (size_t)row * N + col;
        if (resid) val += resid[o];
        if (outf) outf[o] = val;
        if (outh) outh[o] = (_Float16)val;
      }
    }
  }
}

// ---------------------------------------------------------------------------
// Flash attention: grid (S/128, H), 8 waves x 16 query rows each.
// Per 32-key tile: 4 WMMA for Q·K^T, online softmax, 4 WMMA for P·V.
// ---------------------------------------------------------------------------
__global__ __launch_bounds__(256)
void attn_kernel(const _Float16* __restrict__ q, const _Float16* __restrict__ k,
                 const _Float16* __restrict__ v, const int* __restrict__ mask,
                 _Float16* __restrict__ out) {
  __shared__ __align__(16) _Float16 Vt[DKH * 32];      // V^T tile: 64 dk x 32 keys
  __shared__ __align__(16) _Float16 Pl[8 * 16 * 32];   // per-wave P tiles
  int tid = threadIdx.x, lane = tid & 31, wave = tid >> 5;
  int h = blockIdx.y;
  int q0 = blockIdx.x * 128 + wave * 16;
  int l = lane & 15, hi = (lane >> 4) & 1;

  const _Float16* qbase = q + h * DKH;
  v16h qf0 = load_a_frag(qbase,      DMODEL, q0, lane);  // dk 0..31
  v16h qf1 = load_a_frag(qbase + 32, DMODEL, q0, lane);  // dk 32..63

  v8f o0 = {}, o1 = {}, o2 = {}, o3 = {};
  float mrow[8], lrow[8];
#pragma unroll
  for (int i = 0; i < 8; ++i) { mrow[i] = -3.0e38f; lrow[i] = 0.0f; }

  _Float16* pw = &Pl[wave * 16 * 32];
  const _Float16* kbase = k + h * DKH;

  for (int kb = 0; kb < S_LEN; kb += 32) {
    __syncthreads();  // previous iteration done reading Vt
    {   // stage V^T tile: Vt[dk][key]
      int key = tid >> 3;
      int dc = (tid & 7) * 8;
      v8h vv = *(const v8h*)&v[(size_t)(kb + key) * DMODEL + h * DKH + dc];
#pragma unroll
      for (int i = 0; i < 8; ++i) Vt[(dc + i) * 32 + key] = vv[i];
    }
    __syncthreads();

    // scores = Q·K^T : B-frags straight from row-major K (cols=keys, contiguous dk)
    v16h kf00 = load_b_frag(kbase,      DMODEL, kb,      lane);
    v16h kf01 = load_b_frag(kbase + 32, DMODEL, kb,      lane);
    v16h kf10 = load_b_frag(kbase,      DMODEL, kb + 16, lane);
    v16h kf11 = load_b_frag(kbase + 32, DMODEL, kb + 16, lane);
    v8f c0 = {}, c1 = {};
    c0 = WMMA_F16(qf0, kf00, c0);
    c0 = WMMA_F16(qf1, kf01, c0);
    c1 = WMMA_F16(qf0, kf10, c1);
    c1 = WMMA_F16(qf1, kf11, c1);

    int m0 = mask[kb + l], m1 = mask[kb + 16 + l];
#pragma unroll
    for (int i = 0; i < 8; ++i) {
      float s0 = c0[i] * 0.125f, s1 = c1[i] * 0.125f;   // 1/sqrt(64)
      c0[i] = m0 ? s0 : -1.0e9f;
      c1[i] = m1 ? s1 : -1.0e9f;
    }

    // online softmax: row reductions across the 16 lanes holding one row
#pragma unroll
    for (int i = 0; i < 8; ++i) {
      float t = fmaxf(c0[i], c1[i]);
      t = fmaxf(t, __shfl_xor(t, 1, 32));
      t = fmaxf(t, __shfl_xor(t, 2, 32));
      t = fmaxf(t, __shfl_xor(t, 4, 32));
      t = fmaxf(t, __shfl_xor(t, 8, 32));
      float mnew = fmaxf(mrow[i], t);
      float alpha = __expf(mrow[i] - mnew);
      mrow[i] = mnew;
      float p0 = __expf(c0[i] - mnew);
      float p1 = __expf(c1[i] - mnew);
      c0[i] = p0; c1[i] = p1;
      float rs = p0 + p1;
      rs += __shfl_xor(rs, 1, 32);
      rs += __shfl_xor(rs, 2, 32);
      rs += __shfl_xor(rs, 4, 32);
      rs += __shfl_xor(rs, 8, 32);
      lrow[i] = lrow[i] * alpha + rs;
      o0[i] *= alpha; o1[i] *= alpha; o2[i] *= alpha; o3[i] *= alpha;
    }

    // reshape P (C-layout) -> A-layout through per-wave LDS tile
#pragma unroll
    for (int i = 0; i < 8; ++i) {
      int m = i + 8 * hi;
      pw[m * 32 + l]      = (_Float16)c0[i];
      pw[m * 32 + 16 + l] = (_Float16)c1[i];
    }
    v16h pf  = load_a_frag(pw, 32, 0, lane);
    v16h vf0 = load_b_frag(Vt, 32, 0,  lane);
    v16h vf1 = load_b_frag(Vt, 32, 16, lane);
    v16h vf2 = load_b_frag(Vt, 32, 32, lane);
    v16h vf3 = load_b_frag(Vt, 32, 48, lane);
    o0 = WMMA_F16(pf, vf0, o0);
    o1 = WMMA_F16(pf, vf1, o1);
    o2 = WMMA_F16(pf, vf2, o2);
    o3 = WMMA_F16(pf, vf3, o3);
  }

#pragma unroll
  for (int i = 0; i < 8; ++i) {
    float inv = 1.0f / lrow[i];
    size_t row = q0 + i + 8 * hi;
    out[row * DMODEL + h * DKH +  0 + l] = (_Float16)(o0[i] * inv);
    out[row * DMODEL + h * DKH + 16 + l] = (_Float16)(o1[i] * inv);
    out[row * DMODEL + h * DKH + 32 + l] = (_Float16)(o2[i] * inv);
    out[row * DMODEL + h * DKH + 48 + l] = (_Float16)(o3[i] * inv);
  }
}

// ---------------------------------------------------------------------------
extern "C" void kernel_launch(void* const* d_in, const int* in_sizes, int n_in,
                              void* d_out, int out_size, void* d_ws, size_t ws_size,
                              hipStream_t stream) {
  (void)in_sizes; (void)n_in; (void)out_size; (void)ws_size;
  const float* x    = (const float*)d_in[0];
  const int*   mask = (const int*)d_in[1];
  const float* wq   = (const float*)d_in[2];
  const float* bq   = (const float*)d_in[3];
  const float* wk   = (const float*)d_in[4];
  const float* bk   = (const float*)d_in[5];
  const float* wv   = (const float*)d_in[6];
  const float* bv   = (const float*)d_in[7];
  const float* wo   = (const float*)d_in[8];
  const float* bo   = (const float*)d_in[9];
  const float* w1   = (const float*)d_in[10];
  const float* b1   = (const float*)d_in[11];
  const float* w2   = (const float*)d_in[12];
  const float* b2   = (const float*)d_in[13];
  const float* ln1a = (const float*)d_in[14];
  const float* ln1b = (const float*)d_in[15];
  const float* ln2a = (const float*)d_in[16];
  const float* ln2b = (const float*)d_in[17];
  float* outp = (float*)d_out;

  char* ws = (char*)d_ws;
  size_t off = 0;
  auto take = [&](size_t bytes) -> char* {
    char* p = ws + off;
    off += (bytes + 255) & ~(size_t)255;
    return p;
  };
  const size_t SD = (size_t)S_LEN * DMODEL;
  _Float16* h1  = (_Float16*)take(SD * 2);
  _Float16* qh  = (_Float16*)take(SD * 2);
  _Float16* kh  = (_Float16*)take(SD * 2);
  _Float16* vh  = (_Float16*)take(SD * 2);
  _Float16* ah  = (_Float16*)take(SD * 2);
  float*    x1  = (float*)   take(SD * 4);
  _Float16* h2  = (_Float16*)take(SD * 2);
  _Float16* ff1 = (_Float16*)take((size_t)S_LEN * FFDIM * 2);
  _Float16* wqT = (_Float16*)take((size_t)DMODEL * DMODEL * 2);
  _Float16* wkT = (_Float16*)take((size_t)DMODEL * DMODEL * 2);
  _Float16* wvT = (_Float16*)take((size_t)DMODEL * DMODEL * 2);
  _Float16* woT = (_Float16*)take((size_t)DMODEL * DMODEL * 2);
  _Float16* w1T = (_Float16*)take((size_t)DMODEL * FFDIM * 2);
  _Float16* w2T = (_Float16*)take((size_t)FFDIM * DMODEL * 2);

  // 1) weights -> transposed f16
  int nDD = DMODEL * DMODEL, nDF = DMODEL * FFDIM;
  transpose_cast_kernel<<<(nDD + 255) / 256, 256, 0, stream>>>(wq, wqT, DMODEL, DMODEL);
  transpose_cast_kernel<<<(nDD + 255) / 256, 256, 0, stream>>>(wk, wkT, DMODEL, DMODEL);
  transpose_cast_kernel<<<(nDD + 255) / 256, 256, 0, stream>>>(wv, wvT, DMODEL, DMODEL);
  transpose_cast_kernel<<<(nDD + 255) / 256, 256, 0, stream>>>(wo, woT, DMODEL, DMODEL);
  transpose_cast_kernel<<<(nDF + 255) / 256, 256, 0, stream>>>(w1, w1T, DMODEL, FFDIM);
  transpose_cast_kernel<<<(nDF + 255) / 256, 256, 0, stream>>>(w2, w2T, FFDIM, DMODEL);

  dim3 gDD(DMODEL / 128, S_LEN / 128);          // N=512 GEMMs
  dim3 gFF(FFDIM / 128, S_LEN / 128);           // N=2048 GEMM

  // 2) LN1 -> h1 (f16)
  ln_kernel<<<S_LEN, 256, 0, stream>>>(x, ln1a, ln1b, h1);
  // 3) Q/K/V projections (f16 out)
  gemm_kernel<<<gDD, 256, 0, stream>>>(h1, wqT, bq, nullptr, nullptr, qh, S_LEN, DMODEL, DMODEL, 0);
  gemm_kernel<<<gDD, 256, 0, stream>>>(h1, wkT, bk, nullptr, nullptr, kh, S_LEN, DMODEL, DMODEL, 0);
  gemm_kernel<<<gDD, 256, 0, stream>>>(h1, wvT, bv, nullptr, nullptr, vh, S_LEN, DMODEL, DMODEL, 0);
  // 4) flash attention -> ah (f16)
  attn_kernel<<<dim3(S_LEN / 128, NHEAD), 256, 0, stream>>>(qh, kh, vh, mask, ah);
  // 5) output projection + residual -> x1 (f32)
  gemm_kernel<<<gDD, 256, 0, stream>>>(ah, woT, bo, x, x1, nullptr, S_LEN, DMODEL, DMODEL, 0);
  // 6) LN2 -> h2 (f16)
  ln_kernel<<<S_LEN, 256, 0, stream>>>(x1, ln2a, ln2b, h2);
  // 7) FFN up + ReLU -> ff1 (f16)
  gemm_kernel<<<gFF, 256, 0, stream>>>(h2, w1T, b1, nullptr, nullptr, ff1, S_LEN, FFDIM, DMODEL, 1);
  // 8) FFN down + residual -> d_out (f32)
  gemm_kernel<<<gDD, 256, 0, stream>>>(ff1, w2T, b2, x1, outp, nullptr, S_LEN, DMODEL, FFDIM, 0);
}
